// MultiHeadSelfAttention_17970143166757
// MI455X (gfx1250) — matrix operand order repaired and checked by reading, not compiled, not run on hardware
//
#include <hip/hip_runtime.h>

typedef __attribute__((ext_vector_type(16))) __bf16 v16bf;
typedef __attribute__((ext_vector_type(8)))  float  v8f;
typedef int v4i __attribute__((vector_size(16)));   // matches builtin prototype

#define EMBED 1024
#define NHEAD 16
#define HDIM  64
#define SEQ   2048
#define BATCH 4
#define MROWS (BATCH * SEQ)   /* 8192 token rows */

// ---------------- helpers ----------------

__device__ __forceinline__ unsigned short f2bf(float x) {
    unsigned u = __float_as_uint(x);
    u += 0x7fffu + ((u >> 16) & 1u);      // round-to-nearest-even
    return (unsigned short)(u >> 16);
}

union FragU { v16bf v; uint4 q[2]; };

// A-fragment (16x32 bf16): lane<16 -> row M=lane, K={0..7,16..23};
// lane>=16 -> row M=lane-16, K={8..15,24..31}.  Two contiguous 16B loads.
__device__ __forceinline__ v16bf load_frag_a(const unsigned short* buf, int stride,
                                             int row0, int k0) {
    const int lane = threadIdx.x & 31;
    const int lr = lane & 15, half = lane >> 4;
    const unsigned short* p = buf + (size_t)(row0 + lr) * stride + k0 + half * 8;
    FragU f;
    f.q[0] = *(const uint4*)(p);
    f.q[1] = *(const uint4*)(p + 16);
    return f.v;
}

// B-fragment (32x16 bf16): lane<16 -> col N=lane, K={0..7,16..23};
// lane>=16 -> col N=lane-16, K={8..15,24..31}.  Buffer is column-major of B,
// i.e. indexed [n][k] row-major with row stride = `stride`.
__device__ __forceinline__ v16bf load_frag_b(const unsigned short* buf, int stride,
                                             int n0, int k0) {
    const int lane = threadIdx.x & 31;
    const int lr = lane & 15, half = lane >> 4;
    const unsigned short* p = buf + (size_t)(n0 + lr) * stride + k0 + half * 8;
    FragU f;
    f.q[0] = *(const uint4*)(p);
    f.q[1] = *(const uint4*)(p + 16);
    return f.v;
}

__device__ __forceinline__ v8f wmma_bf16(v16bf a, v16bf b, v8f c) {
    return __builtin_amdgcn_wmma_f32_16x16x32_bf16(false, a, false, b,
                                                   (short)0, c, false, false);
}

// 16-byte global -> LDS copy.  Prefer the gfx1250 async-to-LDS pipe
// (GLOBAL_LOAD_ASYNC_TO_LDS_B128, tracked with ASYNCcnt): data never
// touches VGPRs, staging overlaps with WMMA issue.
#if defined(__gfx1250__) && __has_builtin(__builtin_amdgcn_global_load_async_to_lds_b128)
#define HAVE_ASYNC_LDS 1
#else
#define HAVE_ASYNC_LDS 0
#endif

__device__ __forceinline__ void cp16(void* lds_dst, const void* gsrc) {
#if HAVE_ASYNC_LDS
    __builtin_amdgcn_global_load_async_to_lds_b128(
        (__attribute__((address_space(1))) v4i*)gsrc,
        (__attribute__((address_space(3))) v4i*)lds_dst, 0, 0);
#else
    *(uint4*)lds_dst = *(const uint4*)gsrc;
#endif
}

__device__ __forceinline__ void cp16_wait() {
#if HAVE_ASYNC_LDS
#if __has_builtin(__builtin_amdgcn_s_wait_asynccnt)
    __builtin_amdgcn_s_wait_asynccnt(0);
#else
    asm volatile("s_wait_asynccnt 0x0" ::: "memory");
#endif
#endif
}

// ---------------- conversion kernels ----------------

__global__ __launch_bounds__(256) void cvt_bf16_kernel(const float* __restrict__ in,
                                                       unsigned short* __restrict__ out,
                                                       int n) {
    int i = blockIdx.x * 256 + threadIdx.x;
    if (i < n) out[i] = f2bf(in[i]);
}

// W [in=1024, out=1024] fp32 -> Wt [out, in] bf16
__global__ __launch_bounds__(256) void transpose_cvt_kernel(const float* __restrict__ in,
                                                            unsigned short* __restrict__ out) {
    int idx = blockIdx.x * 256 + threadIdx.x;     // over 1024*1024
    int o = idx >> 10;                            // out-dim (row of Wt)
    int i = idx & 1023;                           // in-dim  (col of Wt)
    out[idx] = f2bf(in[(size_t)i * EMBED + o]);
}

// ---------------- tiled WMMA GEMM ----------------
// C[M,1024] = A[M,1024] * Bt^T + bias   (Bt is [1024 out, 1024 in] bf16)
// Block tile 64(M) x 256(N); 8 waves in a 2x4 grid, each wave 32x64
// (2x4 tiles of 16x16).  K staged 64 wide per barrier window (16 WMMA/wave
// between barriers).  A staged in LDS via async-to-LDS, B streamed from
// global (weights are L2-resident: 2MB bf16 vs 192MB L2).
// MODE 0: fp32 out [M,1024]
// MODE 1: bf16 out scattered to [B,H,N,D]   (Q / K layout)
// MODE 2: bf16 out scattered to [B,H,D,N]   (V transposed layout)
template <int MODE>
__global__ __launch_bounds__(256) void gemm_bf16_kernel(
        const unsigned short* __restrict__ A,
        const unsigned short* __restrict__ Bt,
        const float* __restrict__ bias,
        void* __restrict__ Out) {
    __shared__ unsigned short As[64 * 64];        // 8 KB

    const int K = EMBED;
    const int tid = threadIdx.x;
    const int wave = tid >> 5;
    const int wm = wave >> 2, wn = wave & 3;
    const int row_blk = blockIdx.x * 64;
    const int col_blk = blockIdx.y * 256 + wn * 64;

    v8f acc[2][4] = {};

    for (int k0 = 0; k0 < K; k0 += 64) {
        __syncthreads();
        {   // stage A tile 64x64: two 16B async copies per thread
            int r = tid >> 2;
            int c = (tid & 3) * 16;
            const unsigned short* src = &A[(size_t)(row_blk + r) * K + k0 + c];
            cp16(&As[r * 64 + c], src);
            cp16(&As[r * 64 + c + 8], src + 8);
        }
        cp16_wait();
        __syncthreads();

#pragma unroll
        for (int ks = 0; ks < 2; ++ks) {
            const int kk = k0 + ks * 32;
            v16bf bfrag[4];
#pragma unroll
            for (int nt = 0; nt < 4; ++nt)
                bfrag[nt] = load_frag_b(Bt, K, col_blk + nt * 16, kk);
#pragma unroll
            for (int mt = 0; mt < 2; ++mt) {
                v16bf afrag = load_frag_a(As, 64, wm * 32 + mt * 16, ks * 32);
#pragma unroll
                for (int nt = 0; nt < 4; ++nt)
                    acc[mt][nt] = wmma_bf16(afrag, bfrag[nt], acc[mt][nt]);
            }
        }
    }

    // epilogue: C layout -> lane holds col n=lr, rows base+0..7 (base=half*8)
    const int lane = tid & 31;
    const int lr = lane & 15, half = lane >> 4;
#pragma unroll
    for (int mt = 0; mt < 2; ++mt) {
#pragma unroll
        for (int nt = 0; nt < 4; ++nt) {
            int col = col_blk + nt * 16 + lr;
            float bv = bias[col];
#pragma unroll
            for (int i = 0; i < 8; ++i) {
                int row = row_blk + wm * 32 + mt * 16 + half * 8 + i;
                float val = acc[mt][nt][i] + bv;
                if (MODE == 0) {
                    ((float*)Out)[(size_t)row * EMBED + col] = val;
                } else {
                    int b = row >> 11, n = row & (SEQ - 1);
                    int h = col >> 6, d = col & (HDIM - 1);
                    unsigned short o = f2bf(val);
                    if (MODE == 1)
                        ((unsigned short*)Out)[(((size_t)(b * NHEAD + h)) * SEQ + n) * HDIM + d] = o;
                    else
                        ((unsigned short*)Out)[(((size_t)(b * NHEAD + h)) * HDIM + d) * SEQ + n] = o;
                }
            }
        }
    }
}

// ---------------- flash attention ----------------
// Q,K bf16 [B*H, N, 64]; Vt bf16 [B*H, 64, N]; out bf16 [B*N, 1024]
// Block: 256 threads (8 waves), 128 query rows of one (b,h).
// Each wave owns 16 query rows; key loop in chunks of 64.
__global__ __launch_bounds__(256) void attn_kernel(
        const unsigned short* __restrict__ Q,
        const unsigned short* __restrict__ Kb,
        const unsigned short* __restrict__ Vt,
        unsigned short* __restrict__ Outb) {
    __shared__ unsigned short Qs[128 * 64];   // 16 KB
    __shared__ unsigned short Ks[64 * 64];    //  8 KB  [key][d]
    __shared__ unsigned short Vs[64 * 64];    //  8 KB  [d][key]
    __shared__ unsigned short Ps[8][16 * 64]; // 16 KB  per-wave P scratch

    const int bh = blockIdx.y;                // b*H + h
    const int q0 = blockIdx.x * 128;
    const int tid = threadIdx.x;
    const int wave = tid >> 5;
    const int lane = tid & 31;
    const int lr = lane & 15, half = lane >> 4;

    const unsigned short* Qg = Q  + (size_t)bh * SEQ * HDIM;
    const unsigned short* Kg = Kb + (size_t)bh * SEQ * HDIM;
    const unsigned short* Vg = Vt + (size_t)bh * HDIM * SEQ;

    // stage Q block 128x64 (16KB): 4 x 16B async copies per thread
    // (completion covered by the first cp16_wait inside the key loop)
#pragma unroll
    for (int it = 0; it < 4; ++it) {
        int idx = (it * 256 + tid) * 8;
        cp16(&Qs[idx], &Qg[(size_t)q0 * HDIM + idx]);
    }

    const float scale = 0.125f;               // 1/sqrt(64)
    float mrow[8], lrow[8];
#pragma unroll
    for (int i = 0; i < 8; ++i) { mrow[i] = -3.0e38f; lrow[i] = 0.f; }
    v8f accO[4] = {};

    for (int k0 = 0; k0 < SEQ; k0 += 64) {
        __syncthreads();
        {   // stage K chunk [64 keys][64 d] and V chunk [64 d][64 keys]
            int r = tid >> 2;
            int c = (tid & 3) * 16;
            cp16(&Ks[r * 64 + c],     &Kg[(size_t)(k0 + r) * HDIM + c]);
            cp16(&Ks[r * 64 + c + 8], &Kg[(size_t)(k0 + r) * HDIM + c + 8]);
            cp16(&Vs[r * 64 + c],     &Vg[(size_t)r * SEQ + k0 + c]);
            cp16(&Vs[r * 64 + c + 8], &Vg[(size_t)r * SEQ + k0 + c + 8]);
        }
        cp16_wait();
        __syncthreads();

        // S = Q_wave(16x64) @ K_chunk^T(64x64) : 4 n-tiles, 2 k-steps
        v8f s[4] = {};
#pragma unroll
        for (int st = 0; st < 2; ++st) {
            v16bf a = load_frag_a(Qs, 64, wave * 16, st * 32);
#pragma unroll
            for (int nt = 0; nt < 4; ++nt) {
                v16bf b = load_frag_b(Ks, 64, nt * 16, st * 32);
                s[nt] = wmma_bf16(a, b, s[nt]);
            }
        }

        // online softmax over this wave's 16 rows (rows base+0..7, base=half*8)
        float rmax[8];
#pragma unroll
        for (int i = 0; i < 8; ++i) {
#pragma unroll
            for (int nt = 0; nt < 4; ++nt) s[nt][i] *= scale;
            rmax[i] = fmaxf(fmaxf(s[0][i], s[1][i]), fmaxf(s[2][i], s[3][i]));
        }
#pragma unroll
        for (int m = 1; m < 16; m <<= 1)
#pragma unroll
            for (int i = 0; i < 8; ++i)
                rmax[i] = fmaxf(rmax[i], __shfl_xor(rmax[i], m, 32));

        float cf[8], psum[8];
#pragma unroll
        for (int i = 0; i < 8; ++i) {
            float mnew = fmaxf(mrow[i], rmax[i]);
            cf[i] = __expf(mrow[i] - mnew);
            mrow[i] = mnew;
            psum[i] = 0.f;
        }
        // P = exp(S - m) -> per-wave LDS scratch (bf16, A-layout source)
#pragma unroll
        for (int nt = 0; nt < 4; ++nt)
#pragma unroll
            for (int i = 0; i < 8; ++i) {
                float p = __expf(s[nt][i] - mrow[i]);
                psum[i] += p;
                Ps[wave][(half * 8 + i) * 64 + nt * 16 + lr] = f2bf(p);
            }
#pragma unroll
        for (int m = 1; m < 16; m <<= 1)
#pragma unroll
            for (int i = 0; i < 8; ++i)
                psum[i] += __shfl_xor(psum[i], m, 32);
#pragma unroll
        for (int i = 0; i < 8; ++i) lrow[i] = lrow[i] * cf[i] + psum[i];
#pragma unroll
        for (int nt = 0; nt < 4; ++nt)
#pragma unroll
            for (int i = 0; i < 8; ++i) accO[nt][i] *= cf[i];

        // O += P(16x64) @ V_chunk(64x64): 2 k-steps over keys, 4 d-tiles
#pragma unroll
        for (int st = 0; st < 2; ++st) {
            v16bf a = load_frag_a(&Ps[wave][0], 64, 0, st * 32);
#pragma unroll
            for (int nt = 0; nt < 4; ++nt) {
                v16bf b = load_frag_b(Vs, 64, nt * 16, st * 32);
                accO[nt] = wmma_bf16(a, b, accO[nt]);
            }
        }
    }

    // epilogue: write bf16 [token, h*64+d]
    const int b = bh >> 4, h = bh & 15;
#pragma unroll
    for (int nt = 0; nt < 4; ++nt)
#pragma unroll
        for (int i = 0; i < 8; ++i) {
            int n = q0 + wave * 16 + half * 8 + i;
            float val = accO[nt][i] / lrow[i];
            Outb[((size_t)(b * SEQ + n)) * EMBED + h * 64 + nt * 16 + lr] = f2bf(val);
        }
}

// ---------------- launch ----------------

extern "C" void kernel_launch(void* const* d_in, const int* in_sizes, int n_in,
                              void* d_out, int out_size, void* d_ws, size_t ws_size,
                              hipStream_t stream) {
    const float* x  = (const float*)d_in[0];
    const float* Wq = (const float*)d_in[1];
    const float* bq = (const float*)d_in[2];
    const float* Wk = (const float*)d_in[3];
    const float* bk = (const float*)d_in[4];
    const float* Wv = (const float*)d_in[5];
    const float* bv = (const float*)d_in[6];
    const float* Wo = (const float*)d_in[7];
    const float* bo = (const float*)d_in[8];

    char* w = (char*)d_ws;
    auto alloc = [&](size_t bytes) { void* p = (void*)w; w += bytes; return p; };
    const size_t tokBytes = (size_t)MROWS * EMBED * 2;   // 16 MB
    const size_t wBytes   = (size_t)EMBED * EMBED * 2;   //  2 MB

    unsigned short* xbf   = (unsigned short*)alloc(tokBytes);
    unsigned short* WtQ   = (unsigned short*)alloc(wBytes);
    unsigned short* WtK   = (unsigned short*)alloc(wBytes);
    unsigned short* WtV   = (unsigned short*)alloc(wBytes);
    unsigned short* WtO   = (unsigned short*)alloc(wBytes);
    unsigned short* Qb    = (unsigned short*)alloc(tokBytes);
    unsigned short* Kbuf  = (unsigned short*)alloc(tokBytes);
    unsigned short* Vbuf  = (unsigned short*)alloc(tokBytes);
    unsigned short* attnb = (unsigned short*)alloc(tokBytes);
    (void)ws_size; (void)n_in; (void)in_sizes; (void)out_size;

    // 1. casts
    cvt_bf16_kernel<<<(MROWS * EMBED) / 256, 256, 0, stream>>>(x, xbf, MROWS * EMBED);
    transpose_cvt_kernel<<<(EMBED * EMBED) / 256, 256, 0, stream>>>(Wq, WtQ);
    transpose_cvt_kernel<<<(EMBED * EMBED) / 256, 256, 0, stream>>>(Wk, WtK);
    transpose_cvt_kernel<<<(EMBED * EMBED) / 256, 256, 0, stream>>>(Wv, WtV);
    transpose_cvt_kernel<<<(EMBED * EMBED) / 256, 256, 0, stream>>>(Wo, WtO);

    // 2. Q/K/V projections (WMMA GEMMs)
    dim3 ggrid(MROWS / 64, EMBED / 256);
    gemm_bf16_kernel<1><<<ggrid, 256, 0, stream>>>(xbf, WtQ, bq, (void*)Qb);
    gemm_bf16_kernel<1><<<ggrid, 256, 0, stream>>>(xbf, WtK, bk, (void*)Kbuf);
    gemm_bf16_kernel<2><<<ggrid, 256, 0, stream>>>(xbf, WtV, bv, (void*)Vbuf);

    // 3. flash attention
    attn_kernel<<<dim3(SEQ / 128, BATCH * NHEAD), 256, 0, stream>>>(Qb, Kbuf, Vbuf, attnb);

    // 4. output projection -> fp32 d_out
    gemm_bf16_kernel<0><<<ggrid, 256, 0, stream>>>(attnb, WtO, bo, d_out);
}
